// GAT_Trainable_53575422050308
// MI455X (gfx1250) — matrix operand order repaired and checked by reading, not compiled, not run on hardware
//
#include <hip/hip_runtime.h>

#define NND 50000
#define NE  800000
#define NE2 850000      // edges + self loops
#define NG  64
#define NEMB 62
#define LSLOPE 0.2f

typedef __attribute__((ext_vector_type(16))) __bf16 v16bf;
typedef __attribute__((ext_vector_type(8)))  __bf16 v8bf;
typedef __attribute__((ext_vector_type(8)))  float  v8f;

// Native f32 scatter-add (global_atomic_add_f32): workspace is coarse-grained
// device memory, so the "unsafe" (non-CAS) path is the correct one on CDNA5.
__device__ __forceinline__ void atomAddF(float* p, float v) {
  unsafeAtomicAdd(p, v);
}

// ---- monotonic ordered-uint encoding for float atomicMax ----
__device__ __forceinline__ unsigned encF(float x) {
  unsigned u = __float_as_uint(x);
  return u ^ ((u >> 31) ? 0xFFFFFFFFu : 0x80000000u);
}
__device__ __forceinline__ float decF(unsigned u) {
  u ^= (u >> 31) ? 0x80000000u : 0xFFFFFFFFu;
  return __uint_as_float(u);
}
#define ENC_NEG_INF 0x007FFFFFu   // encF(-inf)

__global__ void k_fill_u32(unsigned* p, unsigned v, int n) {
  int i = blockIdx.x * blockDim.x + threadIdx.x;
  if (i < n) p[i] = v;
}

// x = concat(emb[x_ids], degrees) -> bf16 [N,64]
__global__ void k_build_x(const int* __restrict__ xids, const float* __restrict__ degrees,
                          const float* __restrict__ emb, __bf16* __restrict__ xb) {
  int i = blockIdx.x * blockDim.x + threadIdx.x;
  if (i >= NND * 64) return;
  int n = i >> 6, c = i & 63;
  float v = (c < NEMB) ? emb[(size_t)xids[n] * NEMB + c] : degrees[n * 2 + (c - NEMB)];
  xb[i] = (__bf16)v;
}

__global__ void k_f32_to_bf16(const float* __restrict__ s, __bf16* __restrict__ d, int n) {
  int i = blockIdx.x * blockDim.x + threadIdx.x;
  if (i < n) d[i] = (__bf16)s[i];
}

// D[M,KOUT] = A[M,KIN](bf16) @ W[KOUT,KIN](bf16)^T, f32 accumulate.
// One wave computes a 16x64 output strip: A fragment reused across 4 B tiles
// (4 accumulators), K stepped 32 per v_wmma_f32_16x16x32_bf16.
// KIN/KOUT compile-time so all fragment loads are base + immediate offset.
template <int KIN, int KOUT>
__global__ void k_gemm_wmma(const __bf16* __restrict__ A, const __bf16* __restrict__ B,
                            float* __restrict__ D, int M) {
  int wave = (int)((blockIdx.x * blockDim.x + threadIdx.x) >> 5);
  int lane = threadIdx.x & 31;
  int mTiles  = M >> 4;
  int nGroups = KOUT >> 6;                 // groups of 4 column tiles (KOUT % 64 == 0)
  if (wave >= mTiles * nGroups) return;    // wave-uniform: EXEC stays all-1s for WMMA
  int mt = wave % mTiles;
  int ng = wave / mTiles;
  int l16 = lane & 15;
  int grp = lane >> 4;
  // A fragment: row = mt*16 + l16; element i holds K = kt*32 + 16*(i/8) + 8*grp + (i%8)
  const v8bf* arow = (const v8bf*)(A + (size_t)((mt << 4) + l16) * KIN);
  // B fragment j: col = ng*64 + j*16 + l16; element i holds K = kt*32 + 16*(i/8) + 8*grp + (i%8)
  const __bf16* bbase = B + (size_t)((ng << 6) + l16) * KIN;
  v8f acc[4];
#pragma unroll
  for (int j = 0; j < 4; ++j) acc[j] = (v8f){0.f, 0.f, 0.f, 0.f, 0.f, 0.f, 0.f, 0.f};
  for (int kt = 0; kt < KIN / 32; ++kt) {
    v8bf a0 = arow[(kt << 2) + grp];        // K = kt*32 + 8*grp + 0..7
    v8bf a1 = arow[(kt << 2) + 2 + grp];    // K = kt*32 + 16 + 8*grp + 0..7
    v16bf a = __builtin_shufflevector(a0, a1, 0,1,2,3,4,5,6,7,8,9,10,11,12,13,14,15);
#pragma unroll
    for (int j = 0; j < 4; ++j) {
      const v8bf* br = (const v8bf*)(bbase + j * 16 * KIN);  // constant byte offset
      v8bf b0 = br[(kt << 2) + grp];                         // K = kt*32 + 8*grp + 0..7
      v8bf b1 = br[(kt << 2) + 2 + grp];                     // K = kt*32 + 16 + 8*grp + 0..7
      v16bf b = __builtin_shufflevector(b0, b1, 0,1,2,3,4,5,6,7,8,9,10,11,12,13,14,15);
      acc[j] = __builtin_amdgcn_wmma_f32_16x16x32_bf16(false, a, false, b, (short)0,
                                                       acc[j], false, false);
    }
  }
  // D layout: VGPR r -> row M = mt*16 + 8*grp + r, col = ng*64 + j*16 + l16
#pragma unroll
  for (int j = 0; j < 4; ++j) {
    float* drow = D + (size_t)((mt << 4) + (grp << 3)) * KOUT + (ng << 6) + (j << 4) + l16;
#pragma unroll
    for (int r = 0; r < 8; ++r) drow[(size_t)r * KOUT] = acc[j][r];
  }
}

// a_src[n,h] = <h[n,h,:], att_src[h,:]>, a_dst likewise
__global__ void k_att_coeff(const float* __restrict__ h, const float* __restrict__ as,
                            const float* __restrict__ ad, float* __restrict__ asrc,
                            float* __restrict__ adst, int H, int C) {
  int i = blockIdx.x * blockDim.x + threadIdx.x;
  if (i >= NND * H) return;
  int n = i / H, hh = i % H;
  const float* hp = h + (size_t)n * H * C + hh * C;
  float s = 0.f, d = 0.f;
  for (int c = 0; c < C; ++c) {
    float v = hp[c];
    s += v * as[hh * C + c];
    d += v * ad[hh * C + c];
  }
  asrc[i] = s;
  adst[i] = d;
}

__device__ __forceinline__ void edge_sd(int e, const int* src, const int* dst, int& s, int& d) {
  if (e < NE) { s = src[e]; d = dst[e]; }
  else        { s = e - NE; d = e - NE; }   // appended self loops
}

__global__ void k_edge_max(const float* __restrict__ asrc, const float* __restrict__ adst,
                           const int* __restrict__ src, const int* __restrict__ dst,
                           unsigned* __restrict__ emaxE, int H) {
  int e = blockIdx.x * blockDim.x + threadIdx.x;
  if (e >= NE2) return;
  int s, d; edge_sd(e, src, dst, s, d);
  for (int hh = 0; hh < H; ++hh) {
    float v = asrc[s * H + hh] + adst[d * H + hh];
    v = (v > 0.f) ? v : LSLOPE * v;
    atomicMax(&emaxE[d * H + hh], encF(v));
  }
}

__global__ void k_edge_exp(const float* __restrict__ asrc, const float* __restrict__ adst,
                           const int* __restrict__ src, const int* __restrict__ dst,
                           const unsigned* __restrict__ emaxE, float* __restrict__ denom,
                           float* __restrict__ exbuf, int H) {
  int e = blockIdx.x * blockDim.x + threadIdx.x;
  if (e >= NE2) return;
  int s, d; edge_sd(e, src, dst, s, d);
  for (int hh = 0; hh < H; ++hh) {
    float v = asrc[s * H + hh] + adst[d * H + hh];
    v = (v > 0.f) ? v : LSLOPE * v;
    float ex = __expf(v - decF(emaxE[d * H + hh]));
    exbuf[e * H + hh] = ex;
    atomAddF(&denom[d * H + hh], ex);
  }
}

// out[dst] += alpha * h[src], per (edge, head); 64 channels via float4 loads
__global__ void k_edge_msg(const float* __restrict__ h, const int* __restrict__ src,
                           const int* __restrict__ dst, const float* __restrict__ exbuf,
                           const float* __restrict__ denom, float* __restrict__ agg,
                           int H, int C) {
  int i = blockIdx.x * blockDim.x + threadIdx.x;
  if (i >= NE2 * H) return;
  int e = i / H, hh = i % H;
  int s, d; edge_sd(e, src, dst, s, d);
  float alpha = exbuf[e * H + hh] / (denom[d * H + hh] + 1e-16f);
  const float4* hp = (const float4*)(h + (size_t)s * H * C + hh * C);
  float* op = agg + (size_t)d * H * C + hh * C;
  for (int q = 0; q < C / 4; ++q) {
    float4 m = hp[q];
    atomAddF(op + 4 * q + 0, m.x * alpha);
    atomAddF(op + 4 * q + 1, m.y * alpha);
    atomAddF(op + 4 * q + 2, m.z * alpha);
    atomAddF(op + 4 * q + 3, m.w * alpha);
  }
}

__global__ void k_bias_relu(float* __restrict__ x, const float* __restrict__ b, int n, int HC) {
  int i = blockIdx.x * blockDim.x + threadIdx.x;
  if (i >= n) return;
  float v = x[i] + b[i % HC];
  x[i] = v > 0.f ? v : 0.f;
}

__global__ void k_pool(const float* __restrict__ x, const int* __restrict__ batch,
                       float* __restrict__ pooled, float* __restrict__ cnts) {
  int i = blockIdx.x * blockDim.x + threadIdx.x;
  if (i >= NND * 64) return;
  int n = i >> 6, c = i & 63;
  int g = batch[n];
  atomAddF(&pooled[g * 64 + c], x[i]);
  if (c == 0) atomAddF(&cnts[g], 1.0f);
}

__global__ void k_final(const float* __restrict__ pooled, const float* __restrict__ cnts,
                        const float* __restrict__ linW, const float* __restrict__ linb,
                        float* __restrict__ out) {
  int i = blockIdx.x * blockDim.x + threadIdx.x;
  if (i >= NG * 10) return;
  int g = i / 10, cls = i % 10;
  float cnt = cnts[g];
  cnt = cnt > 1.f ? cnt : 1.f;
  float s = 0.f;
  for (int c = 0; c < 64; ++c) s += (pooled[g * 64 + c] / cnt) * linW[cls * 64 + c];
  out[i] = s + linb[cls];
}

extern "C" void kernel_launch(void* const* d_in, const int* in_sizes, int n_in,
                              void* d_out, int out_size, void* d_ws, size_t ws_size,
                              hipStream_t stream) {
  (void)in_sizes; (void)n_in; (void)out_size; (void)ws_size;
  const int*   x_ids    = (const int*)d_in[0];
  const float* degrees  = (const float*)d_in[1];
  const int*   edge_src = (const int*)d_in[2];
  const int*   edge_dst = (const int*)d_in[3];
  const int*   batch    = (const int*)d_in[4];
  const float* emb      = (const float*)d_in[5];
  const float* W1  = (const float*)d_in[6];
  const float* as1 = (const float*)d_in[7];
  const float* ad1 = (const float*)d_in[8];
  const float* b1  = (const float*)d_in[9];
  const float* W2  = (const float*)d_in[10];
  const float* as2 = (const float*)d_in[11];
  const float* ad2 = (const float*)d_in[12];
  const float* b2  = (const float*)d_in[13];
  const float* W3  = (const float*)d_in[14];
  const float* as3 = (const float*)d_in[15];
  const float* ad3 = (const float*)d_in[16];
  const float* b3  = (const float*)d_in[17];
  const float* linW = (const float*)d_in[18];
  const float* linb = (const float*)d_in[19];
  float* out = (float*)d_out;

  // ---- workspace carve-up ----
  char* wsb = (char*)d_ws;
  size_t off = 0;
  auto walloc = [&](size_t bytes) -> void* {
    void* p = wsb + off;
    off += (bytes + 255) & ~(size_t)255;
    return p;
  };
  float*    bufH   = (float*)walloc((size_t)NND * 256 * 4);  // gemm output h
  float*    bufAgg = (float*)walloc((size_t)NND * 256 * 4);  // aggregated output
  __bf16*   bufXB  = (__bf16*)walloc((size_t)NND * 256 * 2); // bf16 activations (gemm A)
  __bf16*   bufWB  = (__bf16*)walloc((size_t)256 * 256 * 2); // bf16 weights (gemm B)
  float*    exbuf  = (float*)walloc((size_t)NE2 * 4 * 4);
  float*    asrc   = (float*)walloc((size_t)NND * 4 * 4);
  float*    adst   = (float*)walloc((size_t)NND * 4 * 4);
  unsigned* emaxE  = (unsigned*)walloc((size_t)NND * 4 * 4);
  float*    denom  = (float*)walloc((size_t)NND * 4 * 4);
  float*    pooled = (float*)walloc((size_t)NG * 64 * 4);
  float*    cnts   = (float*)walloc((size_t)NG * 4);

  const int B = 256;
  auto cdiv = [](long long a, long long b) { return (int)((a + b - 1) / b); };

  // input features -> bf16
  k_build_x<<<cdiv((long long)NND * 64, B), B, 0, stream>>>(x_ids, degrees, emb, bufXB);

  auto layer = [&](const float* W, const float* as, const float* ad, const float* bias,
                   int Kin, int H, int C) {
    int HC = H * C;
    k_f32_to_bf16<<<cdiv((long long)HC * Kin, B), B, 0, stream>>>(W, bufWB, HC * Kin);
    int waves = (NND / 16) * (HC / 64);   // one wave per 16x64 output strip
    int gblk = cdiv((long long)waves * 32, B);
    if (Kin == 64 && HC == 256)
      k_gemm_wmma<64, 256><<<gblk, B, 0, stream>>>(bufXB, bufWB, bufH, NND);
    else if (Kin == 256 && HC == 256)
      k_gemm_wmma<256, 256><<<gblk, B, 0, stream>>>(bufXB, bufWB, bufH, NND);
    else
      k_gemm_wmma<256, 64><<<gblk, B, 0, stream>>>(bufXB, bufWB, bufH, NND);
    k_att_coeff<<<cdiv((long long)NND * H, B), B, 0, stream>>>(bufH, as, ad, asrc, adst, H, C);
    k_fill_u32<<<cdiv((long long)NND * H, B), B, 0, stream>>>(emaxE, ENC_NEG_INF, NND * H);
    k_fill_u32<<<cdiv((long long)NND * H, B), B, 0, stream>>>((unsigned*)denom, 0u, NND * H);
    k_fill_u32<<<cdiv((long long)NND * HC, B), B, 0, stream>>>((unsigned*)bufAgg, 0u, NND * HC);
    k_edge_max<<<cdiv(NE2, B), B, 0, stream>>>(asrc, adst, edge_src, edge_dst, emaxE, H);
    k_edge_exp<<<cdiv(NE2, B), B, 0, stream>>>(asrc, adst, edge_src, edge_dst, emaxE,
                                               denom, exbuf, H);
    k_edge_msg<<<cdiv((long long)NE2 * H, B), B, 0, stream>>>(bufH, edge_src, edge_dst,
                                                              exbuf, denom, bufAgg, H, C);
    k_bias_relu<<<cdiv((long long)NND * HC, B), B, 0, stream>>>(bufAgg, bias, NND * HC, HC);
    k_f32_to_bf16<<<cdiv((long long)NND * HC, B), B, 0, stream>>>(bufAgg, bufXB, NND * HC);
  };

  layer(W1, as1, ad1, b1, 64, 4, 64);    // [N,64]  -> [N,256]
  layer(W2, as2, ad2, b2, 256, 4, 64);   // [N,256] -> [N,256]
  layer(W3, as3, ad3, b3, 256, 1, 64);   // [N,256] -> [N,64] (mean over 1 head == identity)

  // global mean pool + classifier
  k_fill_u32<<<cdiv((long long)NG * 64, B), B, 0, stream>>>((unsigned*)pooled, 0u, NG * 64);
  k_fill_u32<<<1, B, 0, stream>>>((unsigned*)cnts, 0u, NG);
  k_pool<<<cdiv((long long)NND * 64, B), B, 0, stream>>>(bufAgg, batch, pooled, cnts);
  k_final<<<cdiv((long long)NG * 10, B), B, 0, stream>>>(pooled, cnts, linW, linb, out);
}
